// GNN_2972117369038
// MI455X (gfx1250) — matrix operand order repaired and checked by reading, not compiled, not run on hardware
//
#include <hip/hip_runtime.h>

#define N_NODES 30000
#define N_EDGES 240000
#define EMB 300
#define EMB2 600
#define KP1 320            // K of GEMM1 (300 -> pad 320), bf16 agg stride
#define MIDP 608           // mid stride / K of GEMM2 (600 -> pad 608)
#define NP1 640            // W1^T rows pad (600 -> 640 = 10 groups * 4 tiles * 16)
#define NP2 320            // W2^T rows pad (300 -> 320 = 5 groups * 4 tiles * 16)
#define MTILES (N_NODES / 16)   // 1875
#define NG1 10             // N-tile groups in GEMM1 (4 x 16 each, covers 640)
#define NT1_STORE 38       // only tiles covering [0,608) are stored
#define NG2 5              // N-tile groups in GEMM2 (4 x 16 each, covers 320)

typedef __attribute__((ext_vector_type(16))) __bf16          v16bf;
typedef __attribute__((ext_vector_type(8)))  float           v8f;
typedef __attribute__((ext_vector_type(8)))  unsigned short  us8;

union Frag16 { v16bf v; us8 h[2]; unsigned short u[16]; };

static __device__ __forceinline__ unsigned short f2bf(float f) {
    unsigned int u = __float_as_uint(f);
    u += 0x7fffu + ((u >> 16) & 1u);   // round to nearest even
    return (unsigned short)(u >> 16);
}

// ---------------- h0 = x_emb1[x[:,0]] + x_emb2[x[:,1]] ----------------
__global__ __launch_bounds__(256) void k_init_h(const int* __restrict__ x,
        const float* __restrict__ e1, const float* __restrict__ e2,
        float* __restrict__ h) {
    int i = blockIdx.x * 256 + threadIdx.x;
    if (i >= N_NODES * EMB) return;
    int m = i / EMB, c = i - m * EMB;
    h[i] = e1[x[2 * m] * EMB + c] + e2[x[2 * m + 1] * EMB + c];
}

// ---------------- edge scatter: agg[dst] += h[src] + ee ----------------
__global__ __launch_bounds__(256) void k_scatter(const float* __restrict__ h,
        const int* __restrict__ eidx, const int* __restrict__ eattr,
        const float* __restrict__ t1, const float* __restrict__ t2,
        float* __restrict__ agg) {
    int e = blockIdx.x;
    int src, dst, a0, a1;
    if (e < N_EDGES) {
        src = eidx[e]; dst = eidx[N_EDGES + e];
        a0 = eattr[2 * e]; a1 = eattr[2 * e + 1];
    } else {                       // self-loop: bond type 4, direction 0
        src = dst = e - N_EDGES; a0 = 4; a1 = 0;
    }
    const float* hs = h + (size_t)src * EMB;
    const float* r1 = t1 + a0 * EMB;
    const float* r2 = t2 + a1 * EMB;
    float* ad = agg + (size_t)dst * EMB;
    for (int f = threadIdx.x; f < EMB; f += 256)
        unsafeAtomicAdd(&ad[f], hs[f] + r1[f] + r2[f]);
}

// ------- staging: agg f32 [30000x300] -> bf16 [30000x320] (zero-padded K) -------
__global__ __launch_bounds__(256) void k_cvt_agg(const float* __restrict__ agg,
        unsigned short* __restrict__ aggb) {
    int i = blockIdx.x * 256 + threadIdx.x;
    if (i >= N_NODES * KP1) return;
    int m = i / KP1, k = i - m * KP1;
    aggb[i] = (k < EMB) ? f2bf(agg[m * EMB + k]) : (unsigned short)0;
}

// ------- staging: W1 [300x600] -> W1^T bf16 [640x320] (zero-padded both dims) -------
__global__ __launch_bounds__(256) void k_cvt_w1(const float* __restrict__ W,
        unsigned short* __restrict__ Wt) {
    int i = blockIdx.x * 256 + threadIdx.x;
    if (i >= NP1 * KP1) return;
    int n = i / KP1, k = i - n * KP1;
    Wt[i] = (n < EMB2 && k < EMB) ? f2bf(W[(size_t)k * EMB2 + n]) : (unsigned short)0;
}

// ------- staging: W2 [600x300] -> W2^T bf16 [320x608] (zero-padded both dims) -------
__global__ __launch_bounds__(256) void k_cvt_w2(const float* __restrict__ W,
        unsigned short* __restrict__ Wt) {
    int i = blockIdx.x * 256 + threadIdx.x;
    if (i >= NP2 * MIDP) return;
    int n = i / MIDP, k = i - n * MIDP;
    Wt[i] = (n < EMB && k < EMB2) ? f2bf(W[(size_t)k * EMB + n]) : (unsigned short)0;
}

// ------- GEMM1: relu(aggb[30000x320] @ W1t^T + b1) -> bf16 mid[30000x608] -------
__global__ __launch_bounds__(256) void k_gemm1(const unsigned short* __restrict__ A,
        const unsigned short* __restrict__ Wt, const float* __restrict__ bias,
        unsigned short* __restrict__ mid) {
    int wave = threadIdx.x >> 5, lane = threadIdx.x & 31;
    int wid = blockIdx.x * 8 + wave;
    if (wid >= MTILES * NG1) return;
    int mg = wid / NG1, ng = wid - mg * NG1;
    int m0 = mg * 16;
    int mr = lane & 15, half = lane >> 4;
    const unsigned short* arow = A + (size_t)(m0 + mr) * KP1;
    const unsigned short* brow0 = Wt + (size_t)(ng * 64 + mr) * KP1;  // mr == lane&15

    v8f acc[4] = {};
    for (int k0 = 0; k0 < KP1; k0 += 32) {
        if (k0 + 32 < KP1) __builtin_prefetch(arow + k0 + 32, 0, 1);
        Frag16 a;
        a.h[0] = *(const us8*)(arow + k0 + half * 8);        // VGPR 0-3: K = k0+half*8 ..+7
        a.h[1] = *(const us8*)(arow + k0 + 16 + half * 8);   // VGPR 4-7: K = k0+16+half*8 ..+7
        #pragma unroll
        for (int j = 0; j < 4; ++j) {
            const unsigned short* brow = brow0 + (size_t)j * 16 * KP1;
            Frag16 b;
            b.h[0] = *(const us8*)(brow + k0 + half * 8);        // K = k0+half*8 ..+7
            b.h[1] = *(const us8*)(brow + k0 + 16 + half * 8);   // K = k0+16+half*8 ..+7
            acc[j] = __builtin_amdgcn_wmma_f32_16x16x32_bf16(
                false, a.v, false, b.v, (short)0, acc[j], false, false);
        }
    }
    #pragma unroll
    for (int j = 0; j < 4; ++j) {
        int tn = ng * 4 + j;
        if (tn < NT1_STORE) {                // wave-uniform guard
            int n = tn * 16 + mr;
            float bv = (n < EMB2) ? bias[n] : 0.f;
            #pragma unroll
            for (int r = 0; r < 8; ++r) {
                int mm = m0 + half * 8 + r;
                float val = fmaxf(acc[j][r] + bv, 0.f);   // ReLU; padding cols -> 0
                mid[(size_t)mm * MIDP + n] = f2bf(val);
            }
        }
    }
}

// ------- GEMM2: mid[30000x608 bf16] @ W2t^T + b2 -> h2[30000x300] f32 -------
__global__ __launch_bounds__(256) void k_gemm2(const unsigned short* __restrict__ A,
        const unsigned short* __restrict__ Wt, const float* __restrict__ bias,
        float* __restrict__ out) {
    int wave = threadIdx.x >> 5, lane = threadIdx.x & 31;
    int wid = blockIdx.x * 8 + wave;
    if (wid >= MTILES * NG2) return;
    int mg = wid / NG2, ng = wid - mg * NG2;
    int m0 = mg * 16;
    int mr = lane & 15, half = lane >> 4;
    const unsigned short* arow = A + (size_t)(m0 + mr) * MIDP;
    const unsigned short* brow0 = Wt + (size_t)(ng * 64 + mr) * MIDP;

    v8f acc[4] = {};
    for (int k0 = 0; k0 < MIDP; k0 += 32) {
        if (k0 + 32 < MIDP) __builtin_prefetch(arow + k0 + 32, 0, 1);
        Frag16 a;
        a.h[0] = *(const us8*)(arow + k0 + half * 8);
        a.h[1] = *(const us8*)(arow + k0 + 16 + half * 8);
        #pragma unroll
        for (int j = 0; j < 4; ++j) {
            const unsigned short* brow = brow0 + (size_t)j * 16 * MIDP;
            Frag16 b;
            b.h[0] = *(const us8*)(brow + k0 + half * 8);
            b.h[1] = *(const us8*)(brow + k0 + 16 + half * 8);
            acc[j] = __builtin_amdgcn_wmma_f32_16x16x32_bf16(
                false, a.v, false, b.v, (short)0, acc[j], false, false);
        }
    }
    #pragma unroll
    for (int j = 0; j < 4; ++j) {
        int n = (ng * 4 + j) * 16 + mr;
        if (n < EMB) {                        // per-lane guard on stores only
            float bv = bias[n];
            #pragma unroll
            for (int r = 0; r < 8; ++r) {
                int mm = m0 + half * 8 + r;
                out[(size_t)mm * EMB + n] = acc[j][r] + bv;
            }
        }
    }
}

// ---------------- BatchNorm stats: per-feature mean / rsqrt(var+eps) ----------------
__global__ __launch_bounds__(256) void k_bnstats(const float* __restrict__ h2,
        float* __restrict__ mean, float* __restrict__ rstd) {
    int c = blockIdx.x;
    float s = 0.f, ss = 0.f;
    for (int m = threadIdx.x; m < N_NODES; m += 256) {
        float v = h2[(size_t)m * EMB + c];
        s += v; ss += v * v;
    }
    __shared__ float s1[256], s2[256];
    s1[threadIdx.x] = s; s2[threadIdx.x] = ss;
    __syncthreads();
    for (int off = 128; off > 0; off >>= 1) {
        if (threadIdx.x < off) {
            s1[threadIdx.x] += s1[threadIdx.x + off];
            s2[threadIdx.x] += s2[threadIdx.x + off];
        }
        __syncthreads();
    }
    if (threadIdx.x == 0) {
        float mu  = s1[0] * (1.f / N_NODES);
        float var = s2[0] * (1.f / N_NODES) - mu * mu;   // biased variance
        mean[c] = mu;
        rstd[c] = rsqrtf(var + 1e-5f);
    }
}

// ---------------- BatchNorm apply (+ optional ReLU) ----------------
__global__ __launch_bounds__(256) void k_bnapply(const float* __restrict__ h2,
        const float* __restrict__ mean, const float* __restrict__ rstd,
        const float* __restrict__ gamma, const float* __restrict__ beta,
        float* __restrict__ out, int do_relu) {
    int i = blockIdx.x * 256 + threadIdx.x;
    if (i >= N_NODES * EMB) return;
    int c = i % EMB;
    float v = (h2[i] - mean[c]) * rstd[c] * gamma[c] + beta[c];
    if (do_relu) v = fmaxf(v, 0.f);
    out[i] = v;
}

extern "C" void kernel_launch(void* const* d_in, const int* in_sizes, int n_in,
                              void* d_out, int out_size, void* d_ws, size_t ws_size,
                              hipStream_t stream) {
    const int*   x     = (const int*)d_in[0];
    const int*   eidx  = (const int*)d_in[1];
    const int*   eattr = (const int*)d_in[2];
    const float* xe1   = (const float*)d_in[3];
    const float* xe2   = (const float*)d_in[4];
    const float* e1    = (const float*)d_in[5];   // [5,6,300]
    const float* e2    = (const float*)d_in[6];   // [5,3,300]
    const float* W1    = (const float*)d_in[7];   // [5,300,600]
    const float* b1    = (const float*)d_in[8];   // [5,600]
    const float* W2    = (const float*)d_in[9];   // [5,600,300]
    const float* b2    = (const float*)d_in[10];  // [5,300]
    const float* gamma = (const float*)d_in[11];  // [5,300]
    const float* beta  = (const float*)d_in[12];  // [5,300]

    char* ws = (char*)d_ws;
    float*          h    = (float*)(ws + 0);                    //  36,000,000 B
    float*          agg  = (float*)(ws + 36000000);             //  36,000,000 B (also h2)
    unsigned short* mid  = (unsigned short*)(ws + 72000000);    //  36,480,000 B
    unsigned short* aggb = (unsigned short*)(ws + 108480000);   //  19,200,000 B
    unsigned short* w1t  = (unsigned short*)(ws + 127680000);   //     409,600 B
    unsigned short* w2t  = (unsigned short*)(ws + 128089600);   //     389,120 B
    float*          mean = (float*)(ws + 128478720);            //       1,216 B
    float*          rstd = (float*)(ws + 128480000);            //       1,216 B

    const int elemBlocks = (N_NODES * EMB + 255) / 256;
    const int cvtABlocks = (N_NODES * KP1 + 255) / 256;
    const int cvtW1Blocks = (NP1 * KP1 + 255) / 256;
    const int cvtW2Blocks = (NP2 * MIDP + 255) / 256;
    const int g1Blocks = (MTILES * NG1 + 7) / 8;   // 8 waves / block
    const int g2Blocks = (MTILES * NG2 + 7) / 8;

    k_init_h<<<elemBlocks, 256, 0, stream>>>(x, xe1, xe2, h);

    for (int l = 0; l < 5; ++l) {
        hipMemsetAsync(agg, 0, (size_t)N_NODES * EMB * sizeof(float), stream);
        k_scatter<<<N_EDGES + N_NODES, 256, 0, stream>>>(
            h, eidx, eattr, e1 + l * 6 * EMB, e2 + l * 3 * EMB, agg);
        k_cvt_agg<<<cvtABlocks, 256, 0, stream>>>(agg, aggb);
        k_cvt_w1<<<cvtW1Blocks, 256, 0, stream>>>(W1 + (size_t)l * EMB * EMB2, w1t);
        k_cvt_w2<<<cvtW2Blocks, 256, 0, stream>>>(W2 + (size_t)l * EMB2 * EMB, w2t);
        k_gemm1<<<g1Blocks, 256, 0, stream>>>(aggb, w1t, b1 + l * EMB2, mid);
        k_gemm2<<<g2Blocks, 256, 0, stream>>>(mid, w2t, b2 + l * EMB, agg);
        k_bnstats<<<EMB, 256, 0, stream>>>(agg, mean, rstd);
        float* outp = (l == 4) ? (float*)d_out : h;
        k_bnapply<<<elemBlocks, 256, 0, stream>>>(agg, mean, rstd,
            gamma + l * EMB, beta + l * EMB, outp, l < 4 ? 1 : 0);
    }
}
